// SparseCrossAttention_29008209117698
// MI455X (gfx1250) — compile-verified, matrix operand order checked by reading, not run on hardware
//
#include <hip/hip_runtime.h>
#include <math.h>

#define N_TOK 32768
#define NB    16
#define DIMC  512
#define IMGC  768
#define LYC   256
#define NH    8
#define DHC   64

typedef __attribute__((ext_vector_type(16))) __bf16 v16bf;
typedef __attribute__((ext_vector_type(8)))  __bf16 v8bf;
typedef __attribute__((ext_vector_type(8)))  float  v8f;

__device__ __forceinline__ v8f wmma_bf16(v16bf a, v16bf b, v8f c) {
  // D = A(16x32) * B(32x16) + C(16x16 f32), wave32
  return __builtin_amdgcn_wmma_f32_16x16x32_bf16(false, a, false, b, (short)0, c,
                                                 false, false);
}

// Load one 16x32 bf16 fragment row for this lane from LDS.
// Lane (0-15): row r, K = base..base+7 and base+16..base+23 (koff=0)
// Lane (16-31): same row, K shifted by 8 (koff=8)
__device__ __forceinline__ v16bf load_frag(const __bf16* base, int koff) {
  v8bf lo = *(const v8bf*)(base + koff);
  v8bf hi = *(const v8bf*)(base + koff + 16);
  v16bf f;
#pragma unroll
  for (int j = 0; j < 8; ++j) { f[j] = lo[j]; f[j + 8] = hi[j]; }
  return f;
}

__device__ __forceinline__ void split_bf16(float x, __bf16* h, __bf16* l) {
  __bf16 hh = (__bf16)x;
  *h = hh;
  *l = (__bf16)(x - (float)hh);
}

// ---------------------------------------------------------------------------
// offsets[b] = lower_bound(batch_ids, b); offsets[NB] = n
__global__ void offsets_kernel(const int* __restrict__ bid, int n, int nb,
                               int* __restrict__ off) {
  int b = threadIdx.x;
  if (b > nb) return;
  if (b == nb) { off[nb] = n; return; }
  int lo = 0, hi = n;
  while (lo < hi) {
    int mid = (lo + hi) >> 1;
    if (bid[mid] < b) lo = mid + 1; else hi = mid;
  }
  off[b] = lo;
}

// ---------------------------------------------------------------------------
// Row-wise LayerNorm over DIMC=512, one wave32 per row (16 elems/lane).
__global__ __launch_bounds__(256)
void ln_kernel(const float* __restrict__ X, const float* __restrict__ g,
               const float* __restrict__ be, float* __restrict__ Y, int M) {
  int wave = threadIdx.x >> 5, lane = threadIdx.x & 31;
  int row = blockIdx.x * 8 + wave;
  if (row >= M) return;
  const float* x = X + (size_t)row * DIMC;
  float vals[16];
  float s = 0.f;
#pragma unroll
  for (int j = 0; j < 16; ++j) { vals[j] = x[lane + j * 32]; s += vals[j]; }
#pragma unroll
  for (int o = 16; o > 0; o >>= 1) s += __shfl_xor(s, o, 32);
  float mu = s * (1.0f / DIMC);
  float var = 0.f;
#pragma unroll
  for (int j = 0; j < 16; ++j) { float d = vals[j] - mu; var += d * d; }
#pragma unroll
  for (int o = 16; o > 0; o >>= 1) var += __shfl_xor(var, o, 32);
  var *= (1.0f / DIMC);
  float rs = rsqrtf(var + 1e-5f);
  float* yr = Y + (size_t)row * DIMC;
#pragma unroll
  for (int j = 0; j < 16; ++j) {
    int c = lane + j * 32;
    yr[c] = (vals[j] - mu) * rs * g[c] + be[c];
  }
}

// ---------------------------------------------------------------------------
// Generic C[M,512] = A[M,K] * W[512,K]^T + bias, bf16 hi/lo split (3 WMMAs).
// Tile 64x128, 256 threads (8 waves): wave w -> mtile w>>1, 4 n-subtiles.
// EPI: 0=bias  1=bias+exact GELU  2=bias+per-head-transposed V store
//      3=bias+residual(aux)
#define KB   32
#define APAD 8

template <int EPI>
__global__ __launch_bounds__(256)
void gemm_kernel(const float* __restrict__ A, int lda,
                 const float* __restrict__ W, int ldw,
                 const float* __restrict__ bias, float* __restrict__ C,
                 const float* __restrict__ aux, int M, int K) {
  __shared__ __bf16 sAh[64][KB + APAD], sAl[64][KB + APAD];
  __shared__ __bf16 sWh[128][KB + APAD], sWl[128][KB + APAD];
  const int row0 = blockIdx.x * 64;
  const int col0 = blockIdx.y * 128;
  const int tid = threadIdx.x;
  const int wave = tid >> 5, lane = tid & 31;
  const int mt = wave >> 1;
  const int nt0 = (wave & 1) * 4;
  const int r = lane & 15, koff = (lane >> 4) * 8;
  (void)aux;
  v8f acc[4] = {};
  for (int k0 = 0; k0 < K; k0 += KB) {
    __syncthreads();
    for (int idx = tid; idx < 64 * KB; idx += 256) {
      int rr = idx >> 5, cc = idx & 31;
      float x = A[(size_t)(row0 + rr) * lda + k0 + cc];
      split_bf16(x, &sAh[rr][cc], &sAl[rr][cc]);
    }
    for (int idx = tid; idx < 128 * KB; idx += 256) {
      int rr = idx >> 5, cc = idx & 31;
      float x = W[(size_t)(col0 + rr) * ldw + k0 + cc];
      split_bf16(x, &sWh[rr][cc], &sWl[rr][cc]);
    }
    __syncthreads();
    v16bf ah = load_frag(&sAh[mt * 16 + r][0], koff);
    v16bf al = load_frag(&sAl[mt * 16 + r][0], koff);
#pragma unroll
    for (int j = 0; j < 4; ++j) {
      v16bf bh = load_frag(&sWh[(nt0 + j) * 16 + r][0], koff);
      v16bf bl = load_frag(&sWl[(nt0 + j) * 16 + r][0], koff);
      acc[j] = wmma_bf16(ah, bh, acc[j]);
      acc[j] = wmma_bf16(ah, bl, acc[j]);
      acc[j] = wmma_bf16(al, bh, acc[j]);
    }
  }
#pragma unroll
  for (int j = 0; j < 4; ++j) {
    int cg = col0 + (nt0 + j) * 16 + r;
    float bval = bias[cg];
#pragma unroll
    for (int v = 0; v < 8; ++v) {
      int m = row0 + mt * 16 + ((lane >> 4) << 3) + v;
      float val = acc[j][v] + bval;
      if (EPI == 1) val = 0.5f * val * (1.0f + erff(val * 0.70710678118f));
      if (EPI == 3) val += aux[(size_t)m * DIMC + cg];
      if (EPI == 2) {
        int bb = m >> 8, l = m & 255;       // M = NB*LYC, LYC=256
        int h = cg >> 6, d = cg & 63;
        C[(size_t)((bb * NH + h) * DHC + d) * LYC + l] = val;  // Vt[b][h][d][l]
      } else {
        C[(size_t)m * DIMC + cg] = val;
      }
    }
  }
}

// ---------------------------------------------------------------------------
// Attention: per (batch, 64-query tile): S = Q_h K_h^T * 0.125 -> softmax ->
// O_h = P V_h, all heads looped, bf16 hi/lo WMMA, fp32 softmax.
#define QP  72   // 64+8  (bytes-per-row stays 16B aligned)
#define KP2 72
#define SP  264  // 256+8
#define ATTN_LDS (2 * 64 * QP * 2 + 2 * 256 * KP2 * 2 + 64 * SP * 4)  // 159744 B

__global__ __launch_bounds__(256)
void attn_kernel(const float* __restrict__ Q, const float* __restrict__ Kp,
                 const float* __restrict__ Vt, const int* __restrict__ off,
                 float* __restrict__ O) {
  extern __shared__ char smem[];
  __bf16* sQh = (__bf16*)smem;
  __bf16* sQl = sQh + 64 * QP;
  __bf16* sKh = sQl + 64 * QP;
  __bf16* sKl = sKh + 256 * KP2;
  float*  sS  = (float*)(sKl + 256 * KP2);  // 64 x SP f32
  __bf16* sVh = (__bf16*)sS;                // reuse of sS region after softmax
  __bf16* sVl = sVh + 64 * SP;
  __bf16* sAh = sKh;                        // reuse of K region for attn probs
  __bf16* sAl = sKl;

  const int b = blockIdx.y;
  const int base = off[b];
  const int cnt = off[b + 1] - base;
  const int row0 = base + blockIdx.x * 64;
  int rows = cnt - blockIdx.x * 64;
  if (rows <= 0) return;
  if (rows > 64) rows = 64;
  const int tid = threadIdx.x;
  const int wave = tid >> 5, lane = tid & 31;
  const int r = lane & 15, koff = (lane >> 4) * 8;

  for (int h = 0; h < NH; ++h) {
    __syncthreads();  // protect reuse across head iterations
    for (int idx = tid; idx < 64 * 64; idx += 256) {
      int rr = idx >> 6, cc = idx & 63;
      int gr = (rr < rows) ? (row0 + rr) : row0;  // clamp pads to a valid row
      float x = Q[(size_t)gr * DIMC + h * DHC + cc];
      split_bf16(x, &sQh[rr * QP + cc], &sQl[rr * QP + cc]);
    }
    for (int idx = tid; idx < 256 * 64; idx += 256) {
      int rr = idx >> 6, cc = idx & 63;
      float x = Kp[(size_t)(b * LYC + rr) * DIMC + h * DHC + cc];
      split_bf16(x, &sKh[rr * KP2 + cc], &sKl[rr * KP2 + cc]);
    }
    __syncthreads();
    {  // scores: M=64 (4 mtiles), N=256 (16 ntiles), K=64 (2 steps)
      const int mt2 = wave >> 1;
      const int nbase = (wave & 1) * 8;
      v8f acc[8] = {};
      for (int ks = 0; ks < 2; ++ks) {
        v16bf ah = load_frag(sQh + (mt2 * 16 + r) * QP + ks * 32, koff);
        v16bf al = load_frag(sQl + (mt2 * 16 + r) * QP + ks * 32, koff);
#pragma unroll
        for (int j = 0; j < 8; ++j) {
          v16bf bh = load_frag(sKh + ((nbase + j) * 16 + r) * KP2 + ks * 32, koff);
          v16bf bl = load_frag(sKl + ((nbase + j) * 16 + r) * KP2 + ks * 32, koff);
          acc[j] = wmma_bf16(ah, bh, acc[j]);
          acc[j] = wmma_bf16(ah, bl, acc[j]);
          acc[j] = wmma_bf16(al, bh, acc[j]);
        }
      }
#pragma unroll
      for (int j = 0; j < 8; ++j) {
        int n = (nbase + j) * 16 + r;
#pragma unroll
        for (int v = 0; v < 8; ++v) {
          int m = mt2 * 16 + ((lane >> 4) << 3) + v;
          sS[m * SP + n] = acc[j][v] * 0.125f;  // dh^-0.5, dh=64
        }
      }
    }
    __syncthreads();
    if (tid < 64) {  // fp32 softmax, one thread per query row
      float mx = -1e30f;
      for (int l = 0; l < LYC; ++l) mx = fmaxf(mx, sS[tid * SP + l]);
      float sum = 0.f;
      for (int l = 0; l < LYC; ++l) {
        float e = expf(sS[tid * SP + l] - mx);
        sS[tid * SP + l] = e;
        sum += e;
      }
      float inv = 1.0f / sum;
      for (int l = 0; l < LYC; ++l) sS[tid * SP + l] *= inv;
    }
    __syncthreads();
    for (int idx = tid; idx < 64 * 256; idx += 256) {  // probs -> bf16 hi/lo
      int rr = idx >> 8, cc = idx & 255;
      split_bf16(sS[rr * SP + cc], &sAh[rr * SP + cc], &sAl[rr * SP + cc]);
    }
    __syncthreads();
    for (int idx = tid; idx < 64 * 256; idx += 256) {  // stage Vt (overwrites sS)
      int rr = idx >> 8, cc = idx & 255;
      float x = Vt[(size_t)((b * NH + h) * DHC + rr) * LYC + cc];
      split_bf16(x, &sVh[rr * SP + cc], &sVl[rr * SP + cc]);
    }
    __syncthreads();
    {  // O_h = P(64x256) * V(256x64): 4 mtiles x 4 ntiles, K=256 (8 steps)
      const int mt2 = wave >> 1;
      const int nt = (wave & 1) * 2;
      v8f acc[2] = {};
      for (int ks = 0; ks < 8; ++ks) {
        v16bf ah = load_frag(sAh + (mt2 * 16 + r) * SP + ks * 32, koff);
        v16bf al = load_frag(sAl + (mt2 * 16 + r) * SP + ks * 32, koff);
#pragma unroll
        for (int j = 0; j < 2; ++j) {
          v16bf bh = load_frag(sVh + ((nt + j) * 16 + r) * SP + ks * 32, koff);
          v16bf bl = load_frag(sVl + ((nt + j) * 16 + r) * SP + ks * 32, koff);
          acc[j] = wmma_bf16(ah, bh, acc[j]);
          acc[j] = wmma_bf16(ah, bl, acc[j]);
          acc[j] = wmma_bf16(al, bh, acc[j]);
        }
      }
#pragma unroll
      for (int j = 0; j < 2; ++j) {
        int d = (nt + j) * 16 + r;
#pragma unroll
        for (int v = 0; v < 8; ++v) {
          int m = mt2 * 16 + ((lane >> 4) << 3) + v;
          if (m < rows) O[(size_t)(row0 + m) * DIMC + h * DHC + d] = acc[j][v];
        }
      }
    }
  }
}

// ---------------------------------------------------------------------------
extern "C" void kernel_launch(void* const* d_in, const int* in_sizes, int n_in,
                              void* d_out, int out_size, void* d_ws,
                              size_t ws_size, hipStream_t stream) {
  (void)in_sizes; (void)n_in; (void)out_size; (void)ws_size;
  const float* F    = (const float*)d_in[0];
  const float* y    = (const float*)d_in[1];
  const float* ln_g = (const float*)d_in[2];
  const float* ln_b = (const float*)d_in[3];
  const float* Wq   = (const float*)d_in[4];
  const float* Wk   = (const float*)d_in[5];
  const float* Wv   = (const float*)d_in[6];
  const float* bq   = (const float*)d_in[7];
  const float* bk   = (const float*)d_in[8];
  const float* bv   = (const float*)d_in[9];
  const float* Wo   = (const float*)d_in[10];
  const float* bo   = (const float*)d_in[11];
  const float* W1   = (const float*)d_in[12];
  const float* b1   = (const float*)d_in[13];
  const float* W2   = (const float*)d_in[14];
  const float* b2   = (const float*)d_in[15];
  const int*   bid  = (const int*)d_in[16];

  const size_t NW = (size_t)N_TOK * DIMC;        // 16.78M floats
  const size_t YW = (size_t)NB * LYC * DIMC;     // 2.10M floats
  char* wsb = (char*)d_ws;
  int*   off = (int*)wsb;                        // [NB+1]
  float* Xln = (float*)(wsb + 256);              // LN(F)          [N,512]
  float* ym1 = Xln + NW;                         // gelu(y W1^T)   [4096,512]
  float* ym2 = ym1 + YW;                         // ym             [4096,512]
  float* Kp  = ym2 + YW;                         // K              [4096,512]
  float* Vt  = Kp + YW;                          // V^T per head   [B,H,64,256]
  float* Qb  = Vt + YW;                          // Q              [N,512]
  float* Ob  = Qb + NW;                          // attn out       [N,512]
  float* Rb  = Qb;                               // residual sum (reuse Q)

  offsets_kernel<<<1, 32, 0, stream>>>(bid, N_TOK, NB, off);
  ln_kernel<<<N_TOK / 8, 256, 0, stream>>>(F, ln_g, ln_b, Xln, N_TOK);
  // image MLP: gelu(y @ W1^T + b1) @ W2^T + b2
  gemm_kernel<1><<<dim3(64, 4), 256, 0, stream>>>(y, IMGC, W1, IMGC, b1, ym1,
                                                  nullptr, NB * LYC, IMGC);
  gemm_kernel<0><<<dim3(64, 4), 256, 0, stream>>>(ym1, DIMC, W2, DIMC, b2, ym2,
                                                  nullptr, NB * LYC, DIMC);
  // K/V projections (V stored head-transposed)
  gemm_kernel<0><<<dim3(64, 4), 256, 0, stream>>>(ym2, DIMC, Wk, DIMC, bk, Kp,
                                                  nullptr, NB * LYC, DIMC);
  gemm_kernel<2><<<dim3(64, 4), 256, 0, stream>>>(ym2, DIMC, Wv, DIMC, bv, Vt,
                                                  nullptr, NB * LYC, DIMC);
  // Q projection on LayerNormed tokens
  gemm_kernel<0><<<dim3(512, 4), 256, 0, stream>>>(Xln, DIMC, Wq, DIMC, bq, Qb,
                                                   nullptr, N_TOK, DIMC);
  // cross-attention per (batch, 64-token tile); 2560/64 = 40 tiles max
  attn_kernel<<<dim3(40, NB), 256, ATTN_LDS, stream>>>(Qb, Kp, Vt, off, Ob);
  // output projection + residual, then final LayerNorm -> d_out
  gemm_kernel<3><<<dim3(512, 4), 256, 0, stream>>>(Ob, DIMC, Wo, DIMC, bo, Rb,
                                                   Xln, N_TOK, DIMC);
  ln_kernel<<<N_TOK / 8, 256, 0, stream>>>(Rb, ln_g, ln_b, (float*)d_out, N_TOK);
}
